// AutoCorrelationMultiHead_78048145703110
// MI455X (gfx1250) — matrix operand order repaired and checked by reading, not compiled, not run on hardware
//
#include <hip/hip_runtime.h>
#include <hip/hip_bf16.h>
#include <stdint.h>

// ---------------------------------------------------------------------------
// CDNA5 (gfx1250) AutoCorrelation multi-head pipeline
//   1) Q/K projections: bf16 WMMA GEMM (TDM-staged tiles), store -> (b,h,c,s)
//   2) V projection (only s<64 rows used): same GEMM, two TDM tiles per step
//   3) per-column 2048-pt Stockham FFT cross-corr + bitonic sort + softmax
//      (async global->LDS column loads)
//   4) ctx = w @ V : batched bf16 WMMA GEMM (async global->LDS staging)
//   5) out = ctx @ Wo + bo : bf16 WMMA GEMM
// ---------------------------------------------------------------------------

typedef __attribute__((ext_vector_type(16))) __bf16 bf16x16;
typedef __attribute__((ext_vector_type(8)))  float  floatx8;
typedef __attribute__((ext_vector_type(4)))  unsigned int u32x4;
typedef __attribute__((ext_vector_type(8)))  int i32x8;
typedef __attribute__((ext_vector_type(4)))  int i32x4;

#define GAS __attribute__((address_space(1)))
#define LAS __attribute__((address_space(3)))

#if __has_builtin(__builtin_amdgcn_global_load_async_to_lds_b128)
#define HAVE_ASYNC 1
#else
#define HAVE_ASYNC 0
#endif
#if __has_builtin(__builtin_amdgcn_tensor_load_to_lds)
#define HAVE_TDM 1
#else
#define HAVE_TDM 0
#endif

__device__ __forceinline__ void wait_async_all() {
#if __has_builtin(__builtin_amdgcn_s_wait_asynccnt)
  __builtin_amdgcn_s_wait_asynccnt(0);
#else
  asm volatile("s_wait_asynccnt 0x0" ::: "memory");
#endif
}

__device__ __forceinline__ void wait_tensor_all() {
#if __has_builtin(__builtin_amdgcn_s_wait_tensorcnt)
  __builtin_amdgcn_s_wait_tensorcnt(0);
#else
  asm volatile("s_wait_tensorcnt 0x0" ::: "memory");
#endif
}

// 16-byte global -> LDS copy (async when available)
__device__ __forceinline__ void cp_async16(const float* g, float* l) {
#if HAVE_ASYNC
  __builtin_amdgcn_global_load_async_to_lds_b128((GAS i32x4*)g, (LAS i32x4*)l,
                                                 0, 0);
#else
  *(float4*)l = *(const float4*)g;
#endif
}

#if HAVE_TDM
// TDM 2D tile load (f32 elements), with LDS row padding.
// tileX = elements per row, tileY = rows, strideX = global row stride (elems),
// padIntervalCode: pad after 2^(c+1) dwords, padAmountCode: pad (c+1) dwords.
__device__ __forceinline__ void tdm_load_2d(const void* gsrc, const void* ldst,
                                            unsigned tileX, unsigned tileY,
                                            unsigned strideX,
                                            unsigned padIntervalCode,
                                            unsigned padAmountCode) {
  const unsigned long long ga = (unsigned long long)gsrc;
  const unsigned lds = (unsigned)(unsigned long long)(LAS const void*)ldst;
  u32x4 g0;
  g0[0] = 1u;                                        // count=1 (user D#)
  g0[1] = lds;                                       // lds_addr (bytes)
  g0[2] = (unsigned)(ga & 0xffffffffu);              // global_addr lo
  g0[3] = (unsigned)((ga >> 32) & 0x1ffffffu) | (2u << 30);  // hi | type=2
  i32x8 g1;
  const unsigned flags = (2u << 16)                  // data_size = 4 bytes
                       | (1u << 20)                  // pad_enable
                       | (padIntervalCode << 22)
                       | (padAmountCode << 25);
  g1[0] = (int)flags;                                // workgroup_mask=0
  g1[1] = (int)((tileX & 0xffffu) << 16);            // tensor_dim0 lo16
  g1[2] = (int)(((tileX >> 16) & 0xffffu) | ((tileY & 0xffffu) << 16));
  g1[3] = (int)(((tileY >> 16) & 0xffffu) | ((tileX & 0xffffu) << 16));
  g1[4] = (int)(tileY & 0xffffu);                    // tile_dim1, tile_dim2=0
  g1[5] = (int)strideX;                              // tensor_dim0_stride lo32
  g1[6] = 0;
  g1[7] = 0;
  const i32x4 z4 = {0, 0, 0, 0};
#if __clang_major__ >= 23
  const i32x8 z8 = {0, 0, 0, 0, 0, 0, 0, 0};
  __builtin_amdgcn_tensor_load_to_lds(g0, g1, z4, z4, z8, 0);
#else
  __builtin_amdgcn_tensor_load_to_lds(g0, g1, z4, z4, 0);
#endif
}
#endif  // HAVE_TDM

// ---------------- WMMA fragment loaders (contiguous k = 16*half + e, A and B alike) ----------------
// A 16x32 bf16: lane L (<16) row M=L holds K 0-15; lane L+16: K 16-31
__device__ __forceinline__ bf16x16 frag_a_f32(const float* base, int ld) {
  const int lane = threadIdx.x & 31;
  const int r = lane & 15;
  const int half = lane >> 4;
  bf16x16 f;
#pragma unroll
  for (int e = 0; e < 16; ++e) {
    const int k = e + half * 16;
    f[e] = (__bf16)base[r * ld + k];
  }
  return f;
}

// B 32x16 bf16 from natural [k][n] LDS tile: lane n (<16) holds K=0..15 of
// column n; lane n+16 holds K=16..31. baseRow0 points at row k=0, column n0.
__device__ __forceinline__ bf16x16 frag_b_f32(const float* baseRow0, int ld) {
  const int lane = threadIdx.x & 31;
  const int n = lane & 15;
  const int kh = (lane >> 4) * 16;
  bf16x16 f;
#pragma unroll
  for (int e = 0; e < 16; ++e) f[e] = (__bf16)baseRow0[(kh + e) * ld + n];
  return f;
}

// ---------------------------------------------------------------------------
// Main GEMM: C(MxN) = A(MxK) * B(KxN) + bias, fp32 in/out, bf16 WMMA compute.
// MODE 0: plain row-major store
// MODE 1: transposed store into (b,h,c,s) layout
// MODE 2: plain store, A rows remapped m -> (m>>6)*2048 + (m&63) (V proj)
// Tile 128x128x32, 256 threads = 8 waves (2x4), 64x32 per wave, double-buffered
// LDS staged by the Tensor Data Mover (wave 0 issues descriptors).
// ---------------------------------------------------------------------------
#define BM 128
#define BN 128
#define BK 32
#define ALD (BK + 4)   // 36-dword padded A row
#define BLD (BN + 4)   // 132-dword padded B row

template <int MODE>
__global__ __launch_bounds__(256) void gemm_bf16_wmma(
    const float* __restrict__ A, const float* __restrict__ Bw,
    const float* __restrict__ bias, float* __restrict__ C, int M, int N, int K) {
  __shared__ float Asm[2][BM][ALD];   // 36864 B
  __shared__ float Bsm[2][BK][BLD];   // 33792 B

  const int tid = threadIdx.x;
  const int wid = tid >> 5;
  const int m0 = blockIdx.y * BM;
  const int n0 = blockIdx.x * BN;

  // lane-level loader mapping (fallback path)
  const int a_row = tid >> 1;
  const int a_col = (tid & 1) * 16;
  const int b_k = tid >> 3;
  const int b_n = (tid & 7) * 16;
  int a_grow = m0 + a_row;
  if (MODE == 2) a_grow = (a_grow >> 6) * 2048 + (a_grow & 63);
  const float* Aptr = A + (size_t)a_grow * K + a_col;
  const float* Bptr = Bw + (size_t)b_k * N + n0 + b_n;

  auto stage = [&](int buf, int kt) {
#if HAVE_TDM
    if (wid == 0) {
      // B tile: BK rows x BN cols, row stride N; pad 128->132 dwords
      tdm_load_2d(Bw + (size_t)kt * BK * N + n0, &Bsm[buf][0][0], BN, BK, N, 6,
                  3);
      if (MODE == 2) {
        // two contiguous 64-row sub-tiles (row remap crosses batch boundary)
        const int b0 = m0 >> 6;
        tdm_load_2d(A + ((size_t)b0 * 2048) * K + (size_t)kt * BK,
                    &Asm[buf][0][0], BK, 64, K, 4, 3);
        tdm_load_2d(A + ((size_t)(b0 + 1) * 2048) * K + (size_t)kt * BK,
                    &Asm[buf][64][0], BK, 64, K, 4, 3);
      } else {
        // A tile: BM rows x BK cols, row stride K; pad 32->36 dwords
        tdm_load_2d(A + (size_t)m0 * K + (size_t)kt * BK, &Asm[buf][0][0], BK,
                    BM, K, 4, 3);
      }
    }
#else
    const float* ap = Aptr + kt * BK;
    float* al = &Asm[buf][a_row][a_col];
#pragma unroll
    for (int u = 0; u < 4; ++u) cp_async16(ap + 4 * u, al + 4 * u);
    const float* bp = Bptr + (size_t)kt * BK * N;
    float* bl = &Bsm[buf][b_k][b_n];
#pragma unroll
    for (int u = 0; u < 4; ++u) cp_async16(bp + 4 * u, bl + 4 * u);
#endif
  };
  auto stage_wait = [&]() {
#if HAVE_TDM
    if (wid == 0) wait_tensor_all();
#else
    wait_async_all();
#endif
  };

  const int wm = (wid & 1) * 64;   // 2 waves along M
  const int wn = (wid >> 1) * 32;  // 4 waves along N

  floatx8 acc[4][2] = {};

  const int nk = K / BK;
  stage(0, 0);
  stage_wait();
  __syncthreads();

  for (int kt = 0; kt < nk; ++kt) {
    const int cur = kt & 1;
    if (kt + 1 < nk) stage(cur ^ 1, kt + 1);  // overlap with compute below
    bf16x16 af[4];
#pragma unroll
    for (int i = 0; i < 4; ++i)
      af[i] = frag_a_f32(&Asm[cur][wm + 16 * i][0], ALD);
#pragma unroll
    for (int j = 0; j < 2; ++j) {
      const bf16x16 bf = frag_b_f32(&Bsm[cur][0][wn + 16 * j], BLD);
#pragma unroll
      for (int i = 0; i < 4; ++i)
        acc[i][j] = __builtin_amdgcn_wmma_f32_16x16x32_bf16(
            false, af[i], false, bf, (short)0, acc[i][j], false, false);
    }
    if (kt + 1 < nk) stage_wait();
    __syncthreads();
  }

  // epilogue: C/D layout -> row = v + (lane>>4)*8, col = lane&15
  const int lane = tid & 31;
  const int half = lane >> 4;
  const int nl = lane & 15;
#pragma unroll
  for (int i = 0; i < 4; ++i) {
#pragma unroll
    for (int j = 0; j < 2; ++j) {
#pragma unroll
      for (int v = 0; v < 8; ++v) {
        const int mm = m0 + wm + 16 * i + v + half * 8;
        const int nn = n0 + wn + 16 * j + nl;
        const float val = acc[i][j][v] + bias[nn];
        if (MODE == 1) {
          // (b,h,c,s): b = mm>>11, h = nn>>6, c = nn&63, s = mm&2047
          const size_t oi =
              ((size_t)(((mm >> 11) * 16 + (nn >> 6)) * 64 + (nn & 63))) * 2048 +
              (mm & 2047);
          C[oi] = val;
        } else {
          C[(size_t)mm * N + nn] = val;
        }
      }
    }
  }
}

// ---------------------------------------------------------------------------
// Per-column (b,h,c): 2048-pt FFT cross-correlation, bitonic sort (descending),
// softmax; write w[(b,h), t, c]. Stockham autosort radix-2, 11 stages, 256 thr.
// ---------------------------------------------------------------------------
__device__ __forceinline__ void fft_stage(const float* __restrict__ sr,
                                          const float* __restrict__ si,
                                          float* __restrict__ dr,
                                          float* __restrict__ di, int l,
                                          int mshift, float dir) {
  const int m = 1 << mshift;
  const float w0 = dir * (-3.14159265358979323846f) / (float)l;
  for (int t = threadIdx.x; t < 1024; t += 256) {
    const int p = t >> mshift;
    const int q = t & (m - 1);
    float sw, cw;
    __sincosf(w0 * (float)p, &sw, &cw);
    const int ia = q + (p << mshift);
    const int ib = ia + 1024;  // + m*l
    const float arv = sr[ia], aiv = si[ia];
    const float brv = sr[ib], biv = si[ib];
    const int oa = q + (p << (mshift + 1));
    dr[oa] = arv + brv;
    di[oa] = aiv + biv;
    const float xr = arv - brv, xi = aiv - biv;
    dr[oa + m] = xr * cw - xi * sw;
    di[oa + m] = xr * sw + xi * cw;
  }
}

__device__ void fft2048(float*& xr, float*& xi, float*& yr, float*& yi,
                        float dir) {
  int l = 1024, mshift = 0;
  while (l >= 1) {
    __syncthreads();
    fft_stage(xr, xi, yr, yi, l, mshift, dir);
    float* t;
    t = xr; xr = yr; yr = t;
    t = xi; xi = yi; yi = t;
    l >>= 1;
    ++mshift;
  }
  __syncthreads();  // result now in xr/xi
}

__global__ __launch_bounds__(256) void autocorr_sort_softmax(
    const float* __restrict__ Qt, const float* __restrict__ Kt,
    float* __restrict__ Wout) {
  __shared__ float s0r[2048], s0i[2048];
  __shared__ float s1r[2048], s1i[2048];
  __shared__ float s2r[2048], s2i[2048];
  __shared__ float red[256];

  const int col = blockIdx.x;  // ((b*16+h)*64 + c)
  const int bh = col >> 6;
  const int c = col & 63;
  const size_t base = (size_t)col * 2048;
  const int tid = threadIdx.x;

  // async-stage q column into LDS
  {
    const float* gq = Qt + base + tid * 8;
    cp_async16(gq, &s0r[tid * 8]);
    cp_async16(gq + 4, &s0r[tid * 8 + 4]);
  }
  for (int i = tid; i < 2048; i += 256) s0i[i] = 0.f;
  wait_async_all();
  __syncthreads();

  // FFT(q): ping-pong s0 <-> s1, result in s1 (11 stages)
  {
    float *xr = s0r, *xi = s0i, *yr = s1r, *yi = s1i;
    fft2048(xr, xi, yr, yi, 1.f);
  }
  // async-stage k column (s0 free again), FFT(k): s0 <-> s2, result in s2
  {
    const float* gk = Kt + base + tid * 8;
    cp_async16(gk, &s0r[tid * 8]);
    cp_async16(gk + 4, &s0r[tid * 8 + 4]);
  }
  for (int i = tid; i < 2048; i += 256) s0i[i] = 0.f;
  wait_async_all();
  __syncthreads();
  {
    float *xr = s0r, *xi = s0i, *yr = s2r, *yi = s2i;
    fft2048(xr, xi, yr, yi, 1.f);
  }
  // G = Fq * conj(Fk) -> s0
  for (int i = tid; i < 2048; i += 256) {
    const float qr = s1r[i], qi = s1i[i];
    const float kr = s2r[i], ki = s2i[i];
    s0r[i] = qr * kr + qi * ki;
    s0i[i] = qi * kr - qr * ki;
  }
  // inverse FFT: s0 <-> s1; corr = real/2048
  float* arr;
  {
    float *xr = s0r, *xi = s0i, *yr = s1r, *yi = s1i;
    fft2048(xr, xi, yr, yi, -1.f);
    arr = xr;
  }
  for (int i = tid; i < 2048; i += 256) arr[i] *= (1.0f / 2048.0f);
  __syncthreads();

  // bitonic sort, descending
  for (int k = 2; k <= 2048; k <<= 1) {
    for (int j = k >> 1; j > 0; j >>= 1) {
      for (int t = tid; t < 2048; t += 256) {
        const int ixj = t ^ j;
        if (ixj > t) {
          const float a = arr[t], b = arr[ixj];
          if (((t & k) == 0) ? (a < b) : (a > b)) {
            arr[t] = b;
            arr[ixj] = a;
          }
        }
      }
      __syncthreads();
    }
  }

  // softmax over sorted values (max = arr[0])
  const float mx = arr[0];
  float part = 0.f;
  for (int i = tid; i < 2048; i += 256) part += __expf(arr[i] - mx);
  red[tid] = part;
  __syncthreads();
  for (int st = 128; st > 0; st >>= 1) {
    if (tid < st) red[tid] += red[tid + st];
    __syncthreads();
  }
  const float inv = 1.0f / red[0];
  for (int i = tid; i < 2048; i += 256)
    Wout[((size_t)bh * 2048 + i) * 64 + c] = __expf(arr[i] - mx) * inv;
}

// ---------------------------------------------------------------------------
// ctx = w @ V : per (b,h), (2048 x 64) @ (64 x 64), 128-row tiles.
// 256 threads = 8 waves (4 along M x 2 along N), 32x32 per wave.
// Tiles staged with global_load_async_to_lds_b128.
// ---------------------------------------------------------------------------
#define AVLD 68
__global__ __launch_bounds__(256) void av_gemm_wmma(
    const float* __restrict__ Wsm, const float* __restrict__ P,
    float* __restrict__ ctx) {
  __shared__ float As2[128][AVLD];  // 34816 B
  __shared__ float Bs2[64][AVLD];   // 17408 B

  const int tid = threadIdx.x;
  const int bh = blockIdx.y;
  const int b = bh >> 4, h = bh & 15;
  const int t0 = blockIdx.x * 128;

  {  // A: w rows (128 x 64)
    const int r = tid >> 1, c0 = (tid & 1) * 32;
    const float* ap = Wsm + ((size_t)bh * 2048 + t0 + r) * 64 + c0;
    float* al = &As2[r][c0];
#pragma unroll
    for (int u = 0; u < 8; ++u) cp_async16(ap + 4 * u, al + 4 * u);
  }
  {  // B natural [i][j]: Bs2[i][j] = V[b,h,i,j] = P[b*64+i][h*64+j]
    const int i = tid >> 2, j0 = (tid & 3) * 16;
    const float* pp = P + ((size_t)(b * 64 + i)) * 1024 + h * 64 + j0;
    float* bl = &Bs2[i][j0];
#pragma unroll
    for (int u = 0; u < 4; ++u) cp_async16(pp + 4 * u, bl + 4 * u);
  }
  wait_async_all();
  __syncthreads();

  const int wid = tid >> 5;
  const int wm = (wid & 3) * 32;
  const int wn = (wid >> 2) * 32;
  floatx8 acc[2][2] = {};

#pragma unroll
  for (int kk = 0; kk < 64; kk += 32) {
    bf16x16 af[2];
#pragma unroll
    for (int i = 0; i < 2; ++i)
      af[i] = frag_a_f32(&As2[wm + 16 * i][kk], AVLD);
#pragma unroll
    for (int j = 0; j < 2; ++j) {
      const bf16x16 bf = frag_b_f32(&Bs2[kk][wn + 16 * j], AVLD);
#pragma unroll
      for (int i = 0; i < 2; ++i)
        acc[i][j] = __builtin_amdgcn_wmma_f32_16x16x32_bf16(
            false, af[i], false, bf, (short)0, acc[i][j], false, false);
    }
  }

  const int lane = tid & 31, half = lane >> 4, nl = lane & 15;
#pragma unroll
  for (int i = 0; i < 2; ++i)
#pragma unroll
    for (int j = 0; j < 2; ++j)
#pragma unroll
      for (int v = 0; v < 8; ++v) {
        const int tt = t0 + wm + 16 * i + v + half * 8;
        const int jj = wn + 16 * j + nl;
        ctx[((size_t)b * 2048 + tt) * 1024 + h * 64 + jj] = acc[i][j][v];
      }
}

// ---------------------------------------------------------------------------
extern "C" void kernel_launch(void* const* d_in, const int* in_sizes, int n_in,
                              void* d_out, int out_size, void* d_ws,
                              size_t ws_size, hipStream_t stream) {
  const float* x  = (const float*)d_in[0];
  const float* Wq = (const float*)d_in[1];
  const float* bq = (const float*)d_in[2];
  const float* Wk = (const float*)d_in[3];
  const float* bk = (const float*)d_in[4];
  const float* Wv = (const float*)d_in[5];
  const float* bv = (const float*)d_in[6];
  const float* Wo = (const float*)d_in[7];
  const float* bo = (const float*)d_in[8];
  float* out = (float*)d_out;

  const int Bb = 8, S = 2048, D = 1024, H = 16, DK = 64;
  const size_t MT = (size_t)Bb * S;  // 16384

  float* ws  = (float*)d_ws;
  float* Qt  = ws;                        // 16384*1024
  float* Kt  = Qt + MT * D;               // 16384*1024
  float* Pv  = Kt + MT * D;               // 512*1024
  float* Wsm = Pv + (size_t)Bb * DK * D;  // 128*2048*64
  float* ctx = Qt;                        // Qt dead after FFT kernel

  dim3 blk(256);

  // Q / K projections (transposed (b,h,c,s) store)
  gemm_bf16_wmma<1><<<dim3(D / BN, MT / BM), blk, 0, stream>>>(x, Wq, bq, Qt,
                                                               (int)MT, D, D);
  gemm_bf16_wmma<1><<<dim3(D / BN, MT / BM), blk, 0, stream>>>(x, Wk, bk, Kt,
                                                               (int)MT, D, D);
  // V projection: only s<64 rows are consumed downstream
  gemm_bf16_wmma<2><<<dim3(D / BN, (Bb * DK) / BM), blk, 0, stream>>>(
      x, Wv, bv, Pv, Bb * DK, D, D);

  // FFT cross-correlation + descending sort + softmax, one column per block
  autocorr_sort_softmax<<<dim3(Bb * H * DK), blk, 0, stream>>>(Qt, Kt, Wsm);

  // ctx = w @ V
  av_gemm_wmma<<<dim3(S / 128, Bb * H), blk, 0, stream>>>(Wsm, Pv, ctx);

  // out = ctx @ Wo + bo
  gemm_bf16_wmma<0><<<dim3(D / BN, MT / BM), blk, 0, stream>>>(ctx, Wo, bo, out,
                                                               (int)MT, D, D);
}
